// SubstituteGroup_22299470201304
// MI455X (gfx1250) — compile-verified, matrix operand order checked by reading, not run on hardware
//
#include <hip/hip_runtime.h>
#include <math.h>

// ---------------------------------------------------------------------------
// Problem constants (match reference)
// ---------------------------------------------------------------------------
#define C_DIM 2048
#define NV    4096
#define NK    400
#define U_DIM 5
#define MSI   10
#define MGM   5
#define DL    512
#define SUB   400
#define EPSN  1e-6f
#define K_P3  2080   // 2063 padded up to a multiple of 32 for the WMMA K-loop

typedef __bf16 bf16_t;
typedef bf16_t v16bf __attribute__((ext_vector_type(16)));
typedef bf16_t v8bf  __attribute__((ext_vector_type(8)));
typedef float  v8f   __attribute__((ext_vector_type(8)));

__device__ __forceinline__ unsigned short f2bf(float f) {
  unsigned int u = __float_as_uint(f);
  u += 0x7FFFu + ((u >> 16) & 1u);  // round-to-nearest-even
  return (unsigned short)(u >> 16);
}

__device__ __forceinline__ uint4 ldg8(const unsigned short* p, bool pred) {
  return pred ? *(const uint4*)p : make_uint4(0u, 0u, 0u, 0u);
}

__device__ __forceinline__ v16bf frag_from_lds(const unsigned short* base) {
  // 16-bit A/B WMMA fragment: two contiguous 8-element chunks at +0 and +16
  // relative to (row*32 + hf*8), per the documented lane layout.
  v8bf lo = *(const v8bf*)base;
  v8bf hi = *(const v8bf*)(base + 16);
  return __builtin_shufflevector(lo, hi, 0, 1, 2, 3, 4, 5, 6, 7, 8, 9, 10, 11,
                                 12, 13, 14, 15);
}

// ---------------------------------------------------------------------------
// WMMA GEMM, NT-only:  A [M,K] bf16 row-major, B [N,K] bf16 row-major
// (B pre-transposed), C [M,N] f32 = act(scale * A.B^T + bias[col]) + res.
//
// Block: 256 threads = 8 wave32 waves, 128x128 C tile, K stepped by 32.
// Waves arranged 4(M) x 2(N); each wave owns 32x64 -> 2 A-frags x 4 B-frags
// = 8 v_wmma_f32_16x16x32_bf16 per K-step. Next K-tile is prefetched into
// registers while the current tile's WMMAs execute.
// EDGE=false: requires M%128==0, N%128==0, K%32==0 -> no bounds checks at
// all (no exec-mask branches around the prefetch loads). This covers every
// FLOP-heavy GEMM in the pipeline.  EDGE=true: fully guarded fallback.
// ---------------------------------------------------------------------------
template <bool EDGE>
__global__ __launch_bounds__(256) void k_gemm_nt(
    const unsigned short* __restrict__ A, const unsigned short* __restrict__ B,
    float* __restrict__ Cout, const float* __restrict__ bias,
    const float* __restrict__ res, int M, int N, int K, float scale, int act) {
  __shared__ unsigned short As[128 * 32];  // [row][k]
  __shared__ unsigned short Bs[128 * 32];  // [n][k]

  const int tid  = threadIdx.x;
  const int lane = tid & 31;
  const int wave = tid >> 5;
  const int wm   = wave & 3;   // 0..3 : 32-row slab
  const int wn   = wave >> 2;  // 0..1 : 64-col slab
  const int hf   = lane >> 4;
  const int l15  = lane & 15;

  const int m0 = blockIdx.y * 128;
  const int n0 = blockIdx.x * 128;

  // cooperative-load mapping: each thread owns one 16-elem (2 x uint4) strip
  const int lr = tid >> 1;        // 0..127 : tile row
  const int lc = (tid & 1) * 16;  // 0 or 16 : k offset

  v8f acc[2][4];
#pragma unroll
  for (int t = 0; t < 2; ++t)
#pragma unroll
    for (int s = 0; s < 4; ++s) acc[t][s] = v8f{};

  const unsigned short* Arow = A + (size_t)(m0 + lr) * K;
  const unsigned short* Brow = B + (size_t)(n0 + lr) * K;

  uint4 a0, a1, b0, b1;
  if constexpr (EDGE) {
    const bool aok = (m0 + lr) < M;
    const bool bok = (n0 + lr) < N;
    a0 = ldg8(Arow + lc, aok && lc < K);
    a1 = ldg8(Arow + lc + 8, aok && lc + 8 < K);
    b0 = ldg8(Brow + lc, bok && lc < K);
    b1 = ldg8(Brow + lc + 8, bok && lc + 8 < K);
  } else {
    a0 = *(const uint4*)(Arow + lc);
    a1 = *(const uint4*)(Arow + lc + 8);
    b0 = *(const uint4*)(Brow + lc);
    b1 = *(const uint4*)(Brow + lc + 8);
  }

  for (int kb = 0; kb < K; kb += 32) {
    // commit prefetched tile to LDS
    *(uint4*)&As[lr * 32 + lc]     = a0;
    *(uint4*)&As[lr * 32 + lc + 8] = a1;
    *(uint4*)&Bs[lr * 32 + lc]     = b0;
    *(uint4*)&Bs[lr * 32 + lc + 8] = b1;
    __syncthreads();

    // issue next tile's global loads; they fly while we do WMMAs
    const int kn = kb + 32;
    if (kn < K) {
      if constexpr (EDGE) {
        const bool aok = (m0 + lr) < M;
        const bool bok = (n0 + lr) < N;
        a0 = ldg8(Arow + kn + lc, aok && kn + lc < K);
        a1 = ldg8(Arow + kn + lc + 8, aok && kn + lc + 8 < K);
        b0 = ldg8(Brow + kn + lc, bok && kn + lc < K);
        b1 = ldg8(Brow + kn + lc + 8, bok && kn + lc + 8 < K);
      } else {
        a0 = *(const uint4*)(Arow + kn + lc);
        a1 = *(const uint4*)(Arow + kn + lc + 8);
        b0 = *(const uint4*)(Brow + kn + lc);
        b1 = *(const uint4*)(Brow + kn + lc + 8);
      }
    }

    v16bf af[2], bf[4];
#pragma unroll
    for (int t = 0; t < 2; ++t)
      af[t] = frag_from_lds(&As[(wm * 32 + t * 16 + l15) * 32 + hf * 8]);
#pragma unroll
    for (int s = 0; s < 4; ++s)
      bf[s] = frag_from_lds(&Bs[(wn * 64 + s * 16 + l15) * 32 + hf * 8]);

#pragma unroll
    for (int t = 0; t < 2; ++t)
#pragma unroll
      for (int s = 0; s < 4; ++s)
        acc[t][s] = __builtin_amdgcn_wmma_f32_16x16x32_bf16(
            false, af[t], false, bf[s], (short)0, acc[t][s], false, false);
    __syncthreads();
  }

  // epilogue: C/D layout -> lane = N within tile, VGPR v = M + 8*(lane>=16)
#pragma unroll
  for (int t = 0; t < 2; ++t) {
#pragma unroll
    for (int s = 0; s < 4; ++s) {
      const int col = n0 + wn * 64 + s * 16 + l15;
      if (!EDGE || col < N) {
#pragma unroll
        for (int v = 0; v < 8; ++v) {
          const int row = m0 + wm * 32 + t * 16 + hf * 8 + v;
          if (!EDGE || row < M) {
            const size_t idx = (size_t)row * N + col;
            float x = acc[t][s][v] * scale;
            if (bias) x += bias[col];
            if (act == 1)
              x = x > 0.f ? x : 0.1f * x;  // leaky(0.1)
            else if (act == 2)
              x = x > 0.f ? x : 0.f;  // relu
            if (res) x += res[idx];    // fused residual
            Cout[idx] = x;
          }
        }
      }
    }
  }
}

// ---------------------------------------------------------------------------
// Tiled f32 -> bf16 transposing convert (for weights):
// src [R, Ccols] f32  ->  dst [Ccols, Rp] bf16, zero-filled for r in [R, Rp).
// ---------------------------------------------------------------------------
__global__ __launch_bounds__(256) void k_convT_f32(
    const float* __restrict__ src, unsigned short* __restrict__ dst, int R,
    int Ccols, int Rp) {
  __shared__ float t[64][65];
  const int r0 = blockIdx.x * 64;
  const int c0 = blockIdx.y * 64;
  const int tx = threadIdx.x & 63;
  const int ty = threadIdx.x >> 6;  // 0..3
#pragma unroll
  for (int j = 0; j < 16; ++j) {
    const int row = ty + j * 4;
    t[row][tx] = (r0 + row < R && c0 + tx < Ccols)
                     ? src[(size_t)(r0 + row) * Ccols + (c0 + tx)]
                     : 0.f;
  }
  __syncthreads();
#pragma unroll
  for (int j = 0; j < 16; ++j) {
    const int row = ty + j * 4;
    if (c0 + row < Ccols && r0 + tx < Rp)
      dst[(size_t)(c0 + row) * Rp + (r0 + tx)] = f2bf(t[tx][row]);
  }
}

// Tiled bf16 -> bf16 transpose (for activation B operands):
// src [R, Ccols] -> dst [Ccols, R]
__global__ __launch_bounds__(256) void k_transp_bf16(
    const unsigned short* __restrict__ src, unsigned short* __restrict__ dst,
    int R, int Ccols) {
  __shared__ unsigned short t[64][65];
  const int r0 = blockIdx.x * 64;
  const int c0 = blockIdx.y * 64;
  const int tx = threadIdx.x & 63;
  const int ty = threadIdx.x >> 6;
#pragma unroll
  for (int j = 0; j < 16; ++j) {
    const int row = ty + j * 4;
    t[row][tx] = (r0 + row < R && c0 + tx < Ccols)
                     ? src[(size_t)(r0 + row) * Ccols + (c0 + tx)]
                     : (unsigned short)0;
  }
  __syncthreads();
#pragma unroll
  for (int j = 0; j < 16; ++j) {
    const int row = ty + j * 4;
    if (c0 + row < Ccols && r0 + tx < R)
      dst[(size_t)(c0 + row) * R + (r0 + tx)] = t[tx][row];
  }
}

// ---------------------------------------------------------------------------
// Fused row softmax -> bf16 probabilities (no f32 write-back of S).
// One block per row: max pass, sum-of-exp pass, bf16 write pass.
// ---------------------------------------------------------------------------
__global__ __launch_bounds__(256) void k_softmax_bf16(
    const float* __restrict__ S, unsigned short* __restrict__ P, int cols) {
  __shared__ float red[256];
  const float* r = S + (size_t)blockIdx.x * cols;
  unsigned short* o = P + (size_t)blockIdx.x * cols;
  float mx = -3.4e38f;
  for (int i = threadIdx.x; i < cols; i += 256) mx = fmaxf(mx, r[i]);
  red[threadIdx.x] = mx;
  __syncthreads();
  for (int off = 128; off > 0; off >>= 1) {
    if (threadIdx.x < off)
      red[threadIdx.x] = fmaxf(red[threadIdx.x], red[threadIdx.x + off]);
    __syncthreads();
  }
  mx = red[0];
  __syncthreads();
  float sum = 0.f;
  for (int i = threadIdx.x; i < cols; i += 256) sum += __expf(r[i] - mx);
  red[threadIdx.x] = sum;
  __syncthreads();
  for (int off = 128; off > 0; off >>= 1) {
    if (threadIdx.x < off) red[threadIdx.x] += red[threadIdx.x + off];
    __syncthreads();
  }
  const float inv = 1.f / red[0];
  for (int i = threadIdx.x; i < cols; i += 256)
    o[i] = f2bf(__expf(r[i] - mx) * inv);
}

// ---------------------------------------------------------------------------
// Torch-style layernorm (unbiased std), emits bf16 for the next WMMA GEMM.
// ---------------------------------------------------------------------------
__global__ __launch_bounds__(256) void k_layernorm_bf16(
    const float* __restrict__ x, const float* __restrict__ alpha,
    const float* __restrict__ beta, unsigned short* __restrict__ out,
    int cols) {
  __shared__ float s1[256], s2[256];
  const float* xr = x + (size_t)blockIdx.x * cols;
  float sum = 0.f, sq = 0.f;
  for (int i = threadIdx.x; i < cols; i += 256) {
    float v = xr[i];
    sum += v;
    sq += v * v;
  }
  s1[threadIdx.x] = sum;
  s2[threadIdx.x] = sq;
  __syncthreads();
  for (int off = 128; off > 0; off >>= 1) {
    if (threadIdx.x < off) {
      s1[threadIdx.x] += s1[threadIdx.x + off];
      s2[threadIdx.x] += s2[threadIdx.x + off];
    }
    __syncthreads();
  }
  const float mu = s1[0] / (float)cols;
  float var = (s2[0] - (float)cols * mu * mu) / (float)(cols - 1);
  var = fmaxf(var, 0.f);
  const float inv = 1.f / (sqrtf(var) + EPSN);
  unsigned short* orow = out + (size_t)blockIdx.x * cols;
  for (int i = threadIdx.x; i < cols; i += 256)
    orow[i] = f2bf(alpha[i] * (xr[i] - mu) * inv + beta[i]);
}

// f32 -> bf16 convert; act: 0 none, 2 relu.
__global__ void k_f32_to_bf16(const float* __restrict__ src,
                              unsigned short* __restrict__ dst, size_t n,
                              int act) {
  size_t i = (size_t)blockIdx.x * blockDim.x + threadIdx.x;
  if (i >= n) return;
  float v = src[i];
  if (act == 2) v = v > 0.f ? v : 0.f;
  dst[i] = f2bf(v);
}

// Build [U, NK, K_P3] bf16 input for the p3 MLP: concat(Kuser, Kscore^T, II, 0)
__global__ void k_build_p3(const float* __restrict__ Kuser,
                           const float* __restrict__ Kscore,
                           const float* __restrict__ II,
                           unsigned short* __restrict__ dst) {
  size_t i = (size_t)blockIdx.x * blockDim.x + threadIdx.x;
  const size_t total = (size_t)U_DIM * NK * K_P3;
  if (i >= total) return;
  const int c = (int)(i % K_P3);
  const size_t t = i / K_P3;
  const int nk = (int)(t % NK);
  const int u = (int)(t / NK);
  float v;
  if (c < C_DIM)
    v = Kuser[(size_t)u * C_DIM + c];
  else if (c < C_DIM + MSI)
    v = Kscore[(size_t)(c - C_DIM) * NK + nk];  // pref = Kscore^T
  else if (c < C_DIM + MSI + MGM)
    v = II[(size_t)nk * MGM + (c - C_DIM - MSI)];
  else
    v = 0.f;
  dst[i] = f2bf(v);
}

// res[s,v] = cnn_b + sum_u cnn_w[u] * outp[u,v,s]   (the 1x1 conv over U)
__global__ void k_final_mix(const float* __restrict__ outp,
                            const float* __restrict__ cw,
                            const float* __restrict__ cb,
                            float* __restrict__ res) {
  size_t i = (size_t)blockIdx.x * blockDim.x + threadIdx.x;
  if (i >= (size_t)SUB * NV) return;
  const int s = (int)(i / NV);
  const int v = (int)(i % NV);
  float acc = cb[0];
#pragma unroll
  for (int u = 0; u < U_DIM; ++u)
    acc += cw[u] * outp[((size_t)u * NV + v) * SUB + s];
  res[i] = acc;
}

// ---------------------------------------------------------------------------
// Host orchestration
// ---------------------------------------------------------------------------
extern "C" void kernel_launch(void* const* d_in, const int* in_sizes, int n_in,
                              void* d_out, int out_size, void* d_ws,
                              size_t ws_size, hipStream_t stream) {
  (void)in_sizes; (void)n_in; (void)out_size; (void)ws_size;

  const float* Vitem = (const float*)d_in[0];
  const float* Kitem = (const float*)d_in[1];
  const float* Kuser = (const float*)d_in[2];
  /* d_in[3] = Vscore : unused by the reference */
  const float* Kscore = (const float*)d_in[4];
  const float* II = (const float*)d_in[5];
  const float* p1_w1 = (const float*)d_in[6];
  const float* p1_b1 = (const float*)d_in[7];
  const float* p1_w2 = (const float*)d_in[8];
  const float* p2_w1 = (const float*)d_in[9];
  const float* p2_b1 = (const float*)d_in[10];
  const float* p2_w2 = (const float*)d_in[11];
  const float* p3_w1 = (const float*)d_in[12];
  const float* p3_b1 = (const float*)d_in[13];
  const float* p3_w2 = (const float*)d_in[14];
  const float* p3_b2 = (const float*)d_in[15];
  const float* p3_w3 = (const float*)d_in[16];
  const float* lin_w = (const float*)d_in[17];
  const float* lin_b = (const float*)d_in[18];
  const float* cnn_w = (const float*)d_in[19];
  const float* cnn_b = (const float*)d_in[20];
  auto bp = [&](int blk, int idx) -> const float* {
    return (const float*)d_in[21 + (blk - 1) * 8 + idx];
  };

  // ---- bump allocator over d_ws ------------------------------------------
  char* ws = (char*)d_ws;
  size_t off = 0;
  auto allocf = [&](size_t elems) -> float* {
    void* p = ws + off;
    off += ((elems * 4) + 255) & ~(size_t)255;
    return (float*)p;
  };
  auto alloch = [&](size_t elems) -> unsigned short* {
    void* p = ws + off;
    off += ((elems * 2) + 255) & ~(size_t)255;
    return (unsigned short*)p;
  };

  float* t0 = allocf((size_t)NV * C_DIM);
  float* Vitem1 = allocf((size_t)NV * C_DIM);
  float* Kitem1 = allocf((size_t)NK * C_DIM);
  float* keyu = allocf((size_t)U_DIM * NK * C_DIM);
  float* q = allocf((size_t)U_DIM * NV * C_DIM);
  float* ffh = allocf((size_t)U_DIM * NV * DL);
  float* S = allocf((size_t)NV * NV);
  float* outp = allocf((size_t)U_DIM * NV * SUB);
  unsigned short* bfA = alloch((size_t)U_DIM * NV * C_DIM);
  unsigned short* bfB = alloch((size_t)K_P3 * C_DIM);  // largest weight (T)
  unsigned short* bfK = alloch((size_t)U_DIM * NK * C_DIM);
  unsigned short* bfV = alloch((size_t)NK * C_DIM);
  unsigned short* bfP = alloch((size_t)NV * NV);
  unsigned short* bfH = alloch((size_t)U_DIM * NV * DL);
  unsigned short* bfI = alloch((size_t)U_DIM * NK * K_P3);
  unsigned short* bfT = alloch((size_t)C_DIM * NV);  // transposed B acts

  auto cdiv = [](size_t a, size_t b) { return (unsigned)((a + b - 1) / b); };
  const float sscale = 1.0f / sqrtf((float)C_DIM);

  // gemm(A[M,K], Bt[N,K]) -> C[M,N]; res: optional fused "+= / residual" src
  auto gemm = [&](const unsigned short* A, const unsigned short* B, float* Cp,
                  const float* bias, const float* res, int M, int N, int K,
                  float scale, int act) {
    dim3 g((N + 127) / 128, (M + 127) / 128);
    const bool clean = (M % 128 == 0) && (N % 128 == 0) && (K % 32 == 0);
    if (clean)
      k_gemm_nt<false>
          <<<g, 256, 0, stream>>>(A, B, Cp, bias, res, M, N, K, scale, act);
    else
      k_gemm_nt<true>
          <<<g, 256, 0, stream>>>(A, B, Cp, bias, res, M, N, K, scale, act);
  };
  // weight f32 [R=Kin, Ccols=Nout] -> bf16 [Nout, Rp] (transposed, K-padded)
  auto convT = [&](const float* s, unsigned short* d, int R, int Ccols,
                   int Rp) {
    dim3 g((Rp + 63) / 64, (Ccols + 63) / 64);
    k_convT_f32<<<g, 256, 0, stream>>>(s, d, R, Ccols, Rp);
  };
  auto transp = [&](const unsigned short* s, unsigned short* d, int R,
                    int Ccols) {
    dim3 g((R + 63) / 64, (Ccols + 63) / 64);
    k_transp_bf16<<<g, 256, 0, stream>>>(s, d, R, Ccols);
  };
  auto conv = [&](const float* s, unsigned short* d, size_t n, int act) {
    k_f32_to_bf16<<<cdiv(n, 256), 256, 0, stream>>>(s, d, n, act);
  };
  auto ln = [&](const float* x, const float* a, const float* b,
                unsigned short* o, int rows) {
    k_layernorm_bf16<<<rows, 256, 0, stream>>>(x, a, b, o, C_DIM);
  };
  auto softmax = [&](const float* Sp, unsigned short* Pp, int rows, int cols) {
    k_softmax_bf16<<<rows, 256, 0, stream>>>(Sp, Pp, cols);
  };

  // ---- proj1: Vitem1 = leaky(Vitem@w1+b1)@w2 ------------------------------
  conv(Vitem, bfA, (size_t)NV * C_DIM, 0);
  convT(p1_w1, bfB, C_DIM, C_DIM, C_DIM);
  gemm(bfA, bfB, t0, p1_b1, nullptr, NV, C_DIM, C_DIM, 1.f, 1);
  conv(t0, bfA, (size_t)NV * C_DIM, 0);
  convT(p1_w2, bfB, C_DIM, C_DIM, C_DIM);
  gemm(bfA, bfB, Vitem1, nullptr, nullptr, NV, C_DIM, C_DIM, 1.f, 0);

  // ---- proj2: Kitem1 ------------------------------------------------------
  conv(Kitem, bfA, (size_t)NK * C_DIM, 0);
  convT(p2_w1, bfB, C_DIM, C_DIM, C_DIM);
  gemm(bfA, bfB, t0, p2_b1, nullptr, NK, C_DIM, C_DIM, 1.f, 1);
  conv(t0, bfA, (size_t)NK * C_DIM, 0);
  convT(p2_w2, bfB, C_DIM, C_DIM, C_DIM);
  gemm(bfA, bfB, Kitem1, nullptr, nullptr, NK, C_DIM, C_DIM, 1.f, 0);

  // ---- p3 MLP -> key_u [U*NK, C] -----------------------------------------
  k_build_p3<<<cdiv((size_t)U_DIM * NK * K_P3, 256), 256, 0, stream>>>(
      Kuser, Kscore, II, bfI);
  convT(p3_w1, bfB, C_DIM + MSI + MGM, C_DIM, K_P3);
  gemm(bfI, bfB, t0, p3_b1, nullptr, U_DIM * NK, C_DIM, K_P3, 1.f, 1);
  conv(t0, bfA, (size_t)U_DIM * NK * C_DIM, 0);
  convT(p3_w2, bfB, C_DIM, C_DIM, C_DIM);
  gemm(bfA, bfB, t0, p3_b2, nullptr, U_DIM * NK, C_DIM, C_DIM, 1.f, 1);
  conv(t0, bfA, (size_t)U_DIM * NK * C_DIM, 0);
  convT(p3_w3, bfB, C_DIM, C_DIM, C_DIM);
  gemm(bfA, bfB, keyu, nullptr, nullptr, U_DIM * NK, C_DIM, C_DIM, 1.f, 0);

  // ---- transformer-block tail (norm2 + FF + fused residual) ---------------
  auto tail = [&](int pb) {
    ln(q, bp(pb, 2), bp(pb, 3), bfA, U_DIM * NV);
    convT(bp(pb, 4), bfB, C_DIM, DL, C_DIM);  // w1 [C,DL] -> [DL,C]
    gemm(bfA, bfB, ffh, bp(pb, 5), nullptr, U_DIM * NV, DL, C_DIM, 1.f, 2);
    conv(ffh, bfH, (size_t)U_DIM * NV * DL, 0);
    convT(bp(pb, 6), bfB, DL, C_DIM, DL);  // w2 [DL,C] -> [C,DL]
    // q = q + (relu(...)@w2 + b2)  -- residual fused into the GEMM epilogue
    gemm(bfH, bfB, q, bp(pb, 7), q, U_DIM * NV, C_DIM, DL, 1.f, 0);
  };

  // ---- block 1: cross-attention (q broadcast over U) ----------------------
  ln(Vitem1, bp(1, 0), bp(1, 1), bfA, NV);        // q2  [NV, C]
  ln(keyu, bp(1, 0), bp(1, 1), bfK, U_DIM * NK);  // k2  [U*NK, C]
  ln(Kitem1, bp(1, 0), bp(1, 1), bfV, NK);        // v2  [NK, C]
  transp(bfV, bfT, NK, C_DIM);                    // v2^T [C, NK]
  for (int u = 0; u < U_DIM; ++u) {
    // scores: A=q2 [NV,C], B=k2_u [NK,C] (already [N,K] form)
    gemm(bfA, bfK + (size_t)u * NK * C_DIM, S, nullptr, nullptr, NV, NK, C_DIM,
         sscale, 0);
    softmax(S, bfP, NV, NK);
    // q_u = Vitem1 + P @ v2   (residual fused)
    gemm(bfP, bfT, q + (size_t)u * NV * C_DIM, nullptr, Vitem1, NV, C_DIM, NK,
         1.f, 0);
  }
  tail(1);

  // ---- blocks 2,3: self-attention over NV=4096 per user -------------------
  auto self_blk = [&](int pb) {
    ln(q, bp(pb, 0), bp(pb, 1), bfA, U_DIM * NV);  // q2 == k2 == v2
    for (int u = 0; u < U_DIM; ++u) {
      const unsigned short* qa = bfA + (size_t)u * NV * C_DIM;
      float* qu = q + (size_t)u * NV * C_DIM;
      // scores: A=qa [NV,C], B=qa [NV,C] ([N,K] form directly)  [clean path]
      gemm(qa, qa, S, nullptr, nullptr, NV, NV, C_DIM, sscale, 0);
      softmax(S, bfP, NV, NV);
      transp(qa, bfT, NV, C_DIM);  // v2^T [C, NV]
      // q_u += P @ qa            (in-place residual fused)      [clean path]
      gemm(bfP, bfT, qu, nullptr, qu, NV, C_DIM, NV, 1.f, 0);
    }
    tail(pb);
  };
  self_blk(2);
  self_blk(3);

  // ---- output projection + 1x1 conv over U --------------------------------
  conv(q, bfA, (size_t)U_DIM * NV * C_DIM, 2 /*relu before lin*/);
  convT(lin_w, bfB, C_DIM, SUB, C_DIM);  // [C,SUB] -> [SUB,C]
  gemm(bfA, bfB, outp, lin_b, nullptr, U_DIM * NV, SUB, C_DIM, 1.f, 0);
  k_final_mix<<<cdiv((size_t)SUB * NV, 256), 256, 0, stream>>>(
      outp, cnn_w, cnn_b, (float*)d_out);
}
